// ShflLSTM_27324581937438
// MI455X (gfx1250) — matrix operand order repaired and checked by reading, not compiled, not run on hardware
//
#include <hip/hip_runtime.h>

// ---------------------------------------------------------------------------
// ShflLSTM for MI455X (gfx1250, wave32, WMMA)
//   BS=8, SEQ=256, D=128, HS=128, gates G=4*HS=512, cells N=BS*SEQ=2048
//
// Key transforms:
//   * gather(x) @ W == gather(x @ W)  -> precompute XW once (0.27 GFLOP),
//     recurrence is only h@U: 68.7 GFLOP over 256 serial steps.
//   * everything fits in L2 (XW 4MB, U 128KB bf16) -> latency bound, so:
//     U kept in VGPRs for the whole loop, h exchanged via LDS,
//     v_wmma_f32_16x16x32_bf16 (fat K, f32 acc) on the critical path.
//   * XW stored gate-interleaved [row][hcol][4] so the per-cell gather is a
//     single global_load_b128 instead of 4 strided b32 loads.
// ---------------------------------------------------------------------------

#define BS   8
#define SEQ  256
#define D    128
#define HS   128
#define G    512           // 4*HS
#define NCELL (BS*SEQ)     // 2048
#define HID  (BS*SEQ*HS)   // 262144 floats per output tensor

typedef __attribute__((ext_vector_type(16))) __bf16 v16bf;
typedef __attribute__((ext_vector_type(8)))  __bf16 v8bf;
typedef __attribute__((ext_vector_type(8)))  float  v8f;

// ---------------------------------------------------------------- math helpers
__device__ __forceinline__ float sigm_f(float x) {
    return 1.0f / (1.0f + __expf(-x));
}
__device__ __forceinline__ float tanh_f(float x) {
    // overflow-safe: exp(2|x|)=inf -> r=1
    float a = fabsf(x);
    float e = __expf(2.0f * a);
    float r = 1.0f - 2.0f / (e + 1.0f);
    return copysignf(r, x);
}
__device__ __forceinline__ unsigned rotl32(unsigned v, int r) {
    return (v << r) | (v >> (32 - r));
}

// threefry2x32 matching jax.random.uniform(jax.random.key(42), (256,256)):
// key=(0,42); counters split in halves: pair i = (i, i+32768); out = [x0;x1]
__device__ __forceinline__ unsigned jax_bits(unsigned e) {
    const unsigned ks0 = 0u, ks1 = 42u, ks2 = 0x1BD11BDAu ^ 0u ^ 42u;
    unsigned i  = e & 32767u;
    unsigned x0 = i + ks0;
    unsigned x1 = (i + 32768u) + ks1;
    const int RA[4] = {13, 15, 26, 6};
    const int RB[4] = {17, 29, 16, 24};
#pragma unroll
    for (int r = 0; r < 4; ++r) { x0 += x1; x1 = rotl32(x1, RA[r]); x1 ^= x0; }
    x0 += ks1; x1 += ks2 + 1u;
#pragma unroll
    for (int r = 0; r < 4; ++r) { x0 += x1; x1 = rotl32(x1, RB[r]); x1 ^= x0; }
    x0 += ks2; x1 += ks0 + 2u;
#pragma unroll
    for (int r = 0; r < 4; ++r) { x0 += x1; x1 = rotl32(x1, RA[r]); x1 ^= x0; }
    x0 += ks0; x1 += ks1 + 3u;
#pragma unroll
    for (int r = 0; r < 4; ++r) { x0 += x1; x1 = rotl32(x1, RB[r]); x1 ^= x0; }
    x0 += ks1; x1 += ks2 + 4u;
#pragma unroll
    for (int r = 0; r < 4; ++r) { x0 += x1; x1 = rotl32(x1, RA[r]); x1 ^= x0; }
    x0 += ks2; x1 += ks0 + 5u;
    return (e < 32768u) ? x0 : x1;
}

// ------------------------------------------------------------- prep kernel 1:
// idx[s][t]: t<s -> rank-scatter argsort of u[s,0..s-1]; t==s -> s; t>s -> 0
__global__ void k_perm(int* __restrict__ idx_g) {
    __shared__ float us[SEQ];
    __shared__ int   ib[SEQ];
    const int s = blockIdx.x;
    const int t = threadIdx.x;
    unsigned e = (unsigned)(s * SEQ + t);
    unsigned bits = jax_bits(e);
    us[t] = __uint_as_float((bits >> 9) | 0x3F800000u) - 1.0f;
    ib[t] = (t == s) ? s : 0;
    __syncthreads();
    if (t < s) {
        float ut = us[t];
        int r = 0;
        for (int j = 0; j < s; ++j) {
            float uj = us[j];
            r += (uj < ut) || (uj == ut && j < t);   // stable rank
        }
        ib[r] = t;                                    // argsort scatter
    }
    __syncthreads();
    idx_g[s * SEQ + t] = ib[t];
}

// ------------------------------------------------------------- prep kernel 2:
// XW gate-interleaved: XW4[(row*HS + hcol)*4 + g] = (x @ W)[row][g*HS + hcol]
__global__ void k_xw(const float* __restrict__ x, const float* __restrict__ W,
                     float* __restrict__ XW4) {
    int o    = blockIdx.x * blockDim.x + threadIdx.x;  // 0 .. 2048*512-1
    int row  = o >> 9;
    int col  = o & (G - 1);
    int gid  = col >> 7;            // gate 0..3
    int hcol = col & (HS - 1);
    const float* xr = x + (size_t)row * D;
    float acc = 0.0f;
#pragma unroll 8
    for (int k = 0; k < D; ++k) acc = fmaf(xr[k], W[(size_t)k * G + col], acc);
    XW4[((size_t)row * HS + hcol) * 4 + gid] = acc;
}

// ------------------------------------------------------------- prep kernel 3:
// UT[n][k] = (bf16) U[k][n]  — column-major U so B-fragments are K-contiguous
__global__ void k_ut(const float* __restrict__ U, __bf16* __restrict__ UT) {
    int o = blockIdx.x * blockDim.x + threadIdx.x;     // 0 .. 512*128-1
    int n = o >> 7;
    int k = o & (HS - 1);
    UT[o] = (__bf16)U[(size_t)k * G + n];
}

// ------------------------------------------------------------- main kernel:
// 128 blocks x 256 threads; block = 16 cells (b fixed, s in [s0,s0+16)).
// wave w owns gate cols [w*64, w*64+64): 4 WMMA tiles x 4 K-chunks.
__global__ __launch_bounds__(256, 1)
void k_lstm(const float* __restrict__ bias, const float* __restrict__ XW4,
            const __bf16* __restrict__ UT, const int* __restrict__ idx_g,
            float* __restrict__ out) {
    __shared__ __align__(16) float  gates[16][G];     // 32 KB
    __shared__ __align__(16) __bf16 hbf[16][HS];      //  4 KB
    __shared__ int idxs[16][SEQ];                     // 16 KB

    const int tid  = threadIdx.x;
    const int wave = tid >> 5;
    const int lane = tid & 31;
    const int gb   = blockIdx.x;
    const int b    = gb >> 4;
    const int s0   = (gb & 15) << 4;

    // preload this block's 16 idx rows into LDS
    for (int o = tid; o < 16 * SEQ; o += 256) {
        int m = o >> 8, t = o & (SEQ - 1);
        idxs[m][t] = idx_g[(s0 + m) * SEQ + t];
    }
    // h starts at zero
    for (int o = tid; o < 16 * HS; o += 256)
        hbf[o >> 7][o & (HS - 1)] = (__bf16)0.0f;

    // --- B fragments: resident in VGPRs for the entire 256-step loop -------
    // B k order chosen here (A matches, not ISA table order): lane<16 -> N=lane,    K = 0..15  (contiguous)
    //                        lane>=16 -> N=lane-16, K = 16..31 (contiguous)
    const int nloc = lane & 15;
    const int hi   = lane >> 4;           // 0 or 1
    v16bf Bfrag[4][4];
#pragma unroll
    for (int tn = 0; tn < 4; ++tn) {
        int n0 = wave * 64 + tn * 16 + nloc;
#pragma unroll
        for (int kc = 0; kc < 4; ++kc) {
            int kb = kc * 32 + hi * 16;
            v8bf lo8 = *(const v8bf*)&UT[(size_t)n0 * HS + kb];
            v8bf hi8 = *(const v8bf*)&UT[(size_t)n0 * HS + kb + 8];
            v16bf f;
#pragma unroll
            for (int j = 0; j < 8; ++j) { f[j] = lo8[j]; f[8 + j] = hi8[j]; }
            Bfrag[tn][kc] = f;
        }
    }

    // --- per-thread state for the elementwise phase ------------------------
    const int hcol = tid & (HS - 1);
    const int mgrp = tid >> 7;            // 0: cells 0..7, 1: cells 8..15
    const float bi  = bias[hcol];
    const float bff = bias[HS + hcol];
    const float bg  = bias[2 * HS + hcol];
    const float bo  = bias[3 * HS + hcol];
    float c_reg[8];
#pragma unroll
    for (int j = 0; j < 8; ++j) c_reg[j] = 0.0f;

    // gate-interleaved gather base for this (b, hcol): rows are 512 floats
    const float* XWb = XW4 + (size_t)b * SEQ * G + (size_t)hcol * 4;
    __syncthreads();

#pragma unroll 1
    for (int t = 0; t < SEQ; ++t) {
        // ---- h(16x128,bf16) @ U(128 x 64cols) via v_wmma_f32_16x16x32_bf16
        v8f acc[4];
#pragma unroll
        for (int tn = 0; tn < 4; ++tn)
#pragma unroll
            for (int r = 0; r < 8; ++r) acc[tn][r] = 0.0f;

        const int mrow   = lane & 15;
        const int khalfA = hi * 16;       // A layout: lanes>=16 take K+16
#pragma unroll
        for (int kc = 0; kc < 4; ++kc) {
            const int kbA = kc * 32 + khalfA;
            v8bf a0 = *(const v8bf*)&hbf[mrow][kbA];        // K = kb .. kb+7
            v8bf a1 = *(const v8bf*)&hbf[mrow][kbA + 8];    // K = kb+8 ..
            v16bf af;
#pragma unroll
            for (int j = 0; j < 8; ++j) { af[j] = a0[j]; af[8 + j] = a1[j]; }
#pragma unroll
            for (int tn = 0; tn < 4; ++tn)
                acc[tn] = __builtin_amdgcn_wmma_f32_16x16x32_bf16(
                    false, af, false, Bfrag[tn][kc], (short)0, acc[tn],
                    false, false);
        }
        // D/C f32 layout: VGPR r -> M = r + 8*hi, N = nloc
#pragma unroll
        for (int tn = 0; tn < 4; ++tn) {
            int col = wave * 64 + tn * 16 + nloc;
#pragma unroll
            for (int r = 0; r < 8; ++r)
                gates[r + hi * 8][col] = acc[tn][r];
        }
        __syncthreads();

        // ---- fused LSTM cell update (8 cells per thread) ------------------
#pragma unroll
        for (int j = 0; j < 8; ++j) {
            const int m  = mgrp * 8 + j;
            const int sm = s0 + m;
            float gi = gates[m][hcol]          + bi;
            float gf = gates[m][HS + hcol]     + bff;
            float gg = gates[m][2 * HS + hcol] + bg;
            float go = gates[m][3 * HS + hcol] + bo;
            if (t <= sm) {                     // mask[s][t]: one b128 gather
                const float4 xr = *(const float4*)(XWb + (size_t)idxs[m][t] * G);
                gi += xr.x;
                gf += xr.y;
                gg += xr.z;
                go += xr.w;
            }
            if (t + 1 <= sm)                   // prefetch next gather row
                __builtin_prefetch(XWb + (size_t)idxs[m][t + 1] * G, 0, 0);

            float i_t = sigm_f(gi);
            float f_t = sigm_f(gf);
            float g_t = tanh_f(gg);
            float o_t = sigm_f(go);
            float c   = fmaf(f_t, c_reg[j], i_t * g_t);
            c_reg[j]  = c;
            float h   = o_t * tanh_f(c);
            hbf[m][hcol] = (__bf16)h;

            if (t == sm)                       // diagonal output ys[t]
                out[((size_t)b * SEQ + t) * HS + hcol] = h;
            if (t == SEQ - 1) {                // final h, c
                out[(size_t)HID     + ((size_t)b * SEQ + sm) * HS + hcol] = h;
                out[(size_t)2 * HID + ((size_t)b * SEQ + sm) * HS + hcol] = c;
            }
        }
        __syncthreads();
    }
}

// ---------------------------------------------------------------------------
extern "C" void kernel_launch(void* const* d_in, const int* in_sizes, int n_in,
                              void* d_out, int out_size, void* d_ws, size_t ws_size,
                              hipStream_t stream) {
    const float* x  = (const float*)d_in[0];   // (8,256,128)
    const float* W  = (const float*)d_in[1];   // (128,512)
    const float* U  = (const float*)d_in[2];   // (128,512)
    const float* bv = (const float*)d_in[3];   // (512,)
    float* out = (float*)d_out;                // hidden_seq | h | c

    // workspace: idx (256KB) | XW4 (4MB) | UT bf16 (128KB)  ~= 4.4 MB
    char* w = (char*)d_ws;
    int*    idx = (int*)w;
    float*  XW4 = (float*)(w + 262144);
    __bf16* UT  = (__bf16*)(w + 262144 + 4194304);

    k_perm<<<SEQ, SEQ, 0, stream>>>(idx);
    k_xw<<<(NCELL * G) / 256, 256, 0, stream>>>(x, W, XW4);
    k_ut<<<(G * HS) / 256, 256, 0, stream>>>(U, UT);
    k_lstm<<<NCELL / 16, 256, 0, stream>>>(bv, XW4, UT, idx, out);
}
